// CaptionDecoder_37314675868238
// MI455X (gfx1250) — matrix operand order repaired and checked by reading, not compiled, not run on hardware
//
#include <hip/hip_runtime.h>
#include <stdint.h>

#define B_   128
#define S_   80
#define T_   28
#define TM1  27
#define V_   8000
#define WD_  1024
#define H_   512
#define G_   2048   // 4*H
#define LI_  1536   // WD + H

typedef __attribute__((ext_vector_type(16))) __bf16 v16bf;
typedef __attribute__((ext_vector_type(8)))  float  v8f;
typedef __attribute__((ext_vector_type(4)))  unsigned int v4u;
typedef __attribute__((ext_vector_type(8)))  unsigned int v8u;

__device__ __forceinline__ unsigned short f2bf(float f) {
  unsigned int u = __float_as_uint(f);
  u += 0x7FFFu + ((u >> 16) & 1u);           // round-to-nearest-even
  return (unsigned short)(u >> 16);
}
__device__ __forceinline__ float sigm(float x) { return 1.0f / (1.0f + expf(-x)); }

// ---------------------------------------------------------------------------
// Generic bf16 WMMA GEMM: C[M x N] = A(MxK bf16) * B(KxN bf16) (+add0 +add1 +addrow)
// Block tile 128x64, K-step 32, 256 threads = 8 waves (4 in M, 2 in N),
// each wave owns a 32x32 region = 2x2 v_wmma_f32_16x16x32_bf16 tiles.
// A tiles are moved global->LDS by the Tensor Data Mover (tensor_load_to_lds,
// TENSORcnt), double-buffered and overlapped with WMMA on the previous tile.
// B tiles go global->VGPR->LDS (transposed to [n][k]) since TDM can't transpose.
// remapB>0: logical row r = t*remapB + b writes to ((b*remapT + t)*N + col).
// Requires: M % 128 == 0, N % 64 == 0, K % 32 == 0 (true for all uses here).
// ---------------------------------------------------------------------------
#define TMT 128
#define TNT 64
#define TKT 32
#define AP  40   // LDS row pitch (ushorts): 32 data + 8 pad = 16+4 DWORDs (TDM pad)
#define BP  40   // LDS row pitch (ushorts) for B tile (stored transposed: [n][k])

union FragAB { v16bf v; uint4 q[2]; };
union U4u    { uint4 q; unsigned short u[8]; };

// Issue one TDM 2D tile load: 128 rows x 32 bf16, row stride = lda elements,
// into LDS at lds_off with 4-DWORD padding every 16 DWORDs (matches AP=40).
__device__ __forceinline__ void tdm_load_a(unsigned int lds_off,
                                           const unsigned short* gaddr,
                                           unsigned int lda_elems) {
  unsigned long long ga = (unsigned long long)(uintptr_t)gaddr;
  v4u g0;
  g0[0] = 1u;                                   // count=1 (valid), user mode
  g0[1] = lds_off;                              // lds_addr (bytes)
  g0[2] = (unsigned int)ga;                     // global_addr[31:0]
  g0[3] = (unsigned int)((ga >> 32) & 0x1FFFFFFu) | (2u << 30); // addr[56:32]|type=2
  v8u g1;
  g1[0] = (1u << 16)                            // data_size = 2 bytes
        | (1u << 20)                            // pad_enable
        | (3u << 22)                            // pad_interval = 16 DWORDs
        | (3u << 25);                           // pad_amount   = 4 DWORDs
  const unsigned int td0 = 1u << 30, td1 = 1u << 30;  // huge dims: never OOB
  g1[1] = (td0 & 0xFFFFu) << 16;                // [63:48]  tensor_dim0 lo
  g1[2] = (td0 >> 16) | ((td1 & 0xFFFFu) << 16);// [79:64] dim0 hi, [95:80] dim1 lo
  g1[3] = (td1 >> 16) | ((unsigned int)TKT << 16); // [111:96] dim1 hi, [127:112] tile_dim0
  g1[4] = (unsigned int)TMT;                    // [143:128] tile_dim1, tile_dim2=0
  g1[5] = lda_elems;                            // [191:160] tensor_dim0_stride lo
  g1[6] = 0u;                                   // stride hi, dim1_stride lo
  g1[7] = 0u;
  asm volatile("tensor_load_to_lds %0, %1" :: "s"(g0), "s"(g1) : "memory");
}

__global__ __launch_bounds__(256) void cd_gemm_bf16(
    const unsigned short* __restrict__ A, int lda,
    const unsigned short* __restrict__ Bm, int ldb,
    float* __restrict__ C,
    const float* __restrict__ add0,
    const float* __restrict__ add1,
    const float* __restrict__ addrow,
    int N, int K, int remapB, int remapT)
{
  __shared__ unsigned short As[2][TMT * AP];
  __shared__ unsigned short Bs[2][TNT * BP];
  const int tid  = threadIdx.x;
  const int lane = tid & 31;
  const int wv   = tid >> 5;
  const int wm   = (wv & 3) << 5;       // wave row offset in tile
  const int wn   = (wv >> 2) << 5;      // wave col offset in tile
  const int m0   = blockIdx.y * TMT;
  const int n0   = blockIdx.x * TNT;
  const int ll   = lane & 15;
  const int lh   = lane >> 4;
  const bool wv0 = (wv == 0);

  // B global->LDS mapping: 32 rows x 64 cols, 8 threads/row (8 bf16 each)
  const int bk = tid >> 3;
  const int bn = (tid & 7) << 3;

  const unsigned short* aTile = A + (size_t)m0 * lda;   // + kc per tile
  const unsigned int asOff[2] = { (unsigned int)(uintptr_t)&As[0][0],
                                  (unsigned int)(uintptr_t)&As[1][0] };

  v8f acc[2][2] = {};
  const int nk = K / TKT;

  // ---- prologue: stage tile 0 ----
  if (wv0) tdm_load_a(asOff[0], aTile, (unsigned int)lda);
  {
    U4u bu; bu.q = *(const uint4*)(Bm + (size_t)bk * ldb + n0 + bn);
#pragma unroll
    for (int j = 0; j < 8; ++j) Bs[0][(bn + j) * BP + bk] = bu.u[j];
  }
  if (wv0) __builtin_amdgcn_s_wait_tensorcnt(0);
  __syncthreads();                         // tile 0 published

  for (int i = 0; i < nk; ++i) {
    const int cur = i & 1, nxt = cur ^ 1;
    const bool more = (i + 1 < nk);
    U4u bu;
    if (more) {                            // kick off tile i+1 while computing i
      const int kc = (i + 1) * TKT;
      if (wv0) tdm_load_a(asOff[nxt], aTile + kc, (unsigned int)lda);
      const unsigned short* bg = Bm + (size_t)(kc + bk) * ldb + n0 + bn;
      bu.q = *(const uint4*)(bg);
      if (i + 2 < nk) __builtin_prefetch(bg + (size_t)TKT * ldb, 0, 1);
    }

    // ---- compute tile i ----
    const unsigned short* Ac = &As[cur][0];
    const unsigned short* Bc = &Bs[cur][0];
    // A frag: lanes 0-15: M=lane, elems 0-7 -> K 0..7, 8-15 -> K 16..23
    //         lanes 16-31: elems 0-7 -> K 8..15, 8-15 -> K 24..31
    FragAB af[2], bfr[2];
#pragma unroll
    for (int mi = 0; mi < 2; ++mi) {
      int row = wm + (mi << 4) + ll;
      int ko  = lh << 3;
      af[mi].q[0] = *(const uint4*)(&Ac[row * AP + ko]);
      af[mi].q[1] = *(const uint4*)(&Ac[row * AP + ko + 16]);
    }
    // B frag: same K order as A: elems 0-7 -> K 8*lh.., 8-15 -> K 16+8*lh.. (Bs is [n][k])
#pragma unroll
    for (int ni = 0; ni < 2; ++ni) {
      int col = wn + (ni << 4) + ll;
      int ko  = lh << 3;
      bfr[ni].q[0] = *(const uint4*)(&Bc[col * BP + ko]);
      bfr[ni].q[1] = *(const uint4*)(&Bc[col * BP + ko + 16]);
    }
#pragma unroll
    for (int mi = 0; mi < 2; ++mi)
#pragma unroll
      for (int ni = 0; ni < 2; ++ni)
        acc[mi][ni] = __builtin_amdgcn_wmma_f32_16x16x32_bf16(
            false, af[mi].v, false, bfr[ni].v, (short)0, acc[mi][ni], false, false);

    __syncthreads();                       // everyone done reading buffers of i
    if (more) {
#pragma unroll
      for (int j = 0; j < 8; ++j) Bs[nxt][(bn + j) * BP + bk] = bu.u[j];
      if (wv0) __builtin_amdgcn_s_wait_tensorcnt(0);   // DMA for i+1 done
      __syncthreads();                     // tile i+1 published
    }
  }

  // epilogue: C/D layout: VGPR j, lanes 0-15 -> M=j, lanes 16-31 -> M=8+j; N=lane&15
#pragma unroll
  for (int mi = 0; mi < 2; ++mi) {
#pragma unroll
    for (int ni = 0; ni < 2; ++ni) {
      int cl = n0 + wn + (ni << 4) + ll;
#pragma unroll
      for (int j = 0; j < 8; ++j) {
        int r = m0 + wm + (mi << 4) + (lh << 3) + j;
        float v = acc[mi][ni][j];
        if (add0)   v += add0[(size_t)r * N + cl];
        if (add1)   v += add1[(size_t)r * N + cl];
        if (addrow) v += addrow[cl];
        size_t off;
        if (remapB > 0) { int b = r % remapB, t = r / remapB;
                          off = ((size_t)b * remapT + t) * N + cl; }
        else            { off = (size_t)r * N + cl; }
        C[off] = v;
      }
    }
  }
}

// out[m] = sum_k W[k*ld + m] * v[k]   (column-reduction GEMV for the linear collapse)
__global__ void cd_colvec(const float* __restrict__ W, int ld,
                          const float* __restrict__ v, int K,
                          float* __restrict__ out, int n) {
  int m = blockIdx.x * blockDim.x + threadIdx.x;
  if (m >= n) return;
  float s = 0.f;
  for (int k = 0; k < K; ++k) s += W[(size_t)k * ld + m] * v[k];
  out[m] = s;
}

// per-batch: scores = enc . t0e, softmax over S, ctx = sum_s w_s * enc_s  (-> bf16)
__global__ __launch_bounds__(256) void cd_attn_ctx(
    const float* __restrict__ enc, const float* __restrict__ t0e,
    unsigned short* __restrict__ ctx_bf) {
  __shared__ float sv[H_];
  __shared__ float sw[S_];
  int b = blockIdx.x, tid = threadIdx.x;
  sv[tid] = t0e[tid]; sv[tid + 256] = t0e[tid + 256];
  __syncthreads();
  const float* eb = enc + (size_t)b * S_ * H_;
  if (tid < S_) {
    const float* er = eb + (size_t)tid * H_;
    float s = 0.f;
    for (int d = 0; d < H_; ++d) s += er[d] * sv[d];
    sw[tid] = s;
  }
  __syncthreads();
  if (tid == 0) {
    float mx = sw[0];
    for (int s = 1; s < S_; ++s) mx = fmaxf(mx, sw[s]);
    float sum = 0.f;
    for (int s = 0; s < S_; ++s) { float e = expf(sw[s] - mx); sw[s] = e; sum += e; }
    float inv = 1.0f / sum;
    for (int s = 0; s < S_; ++s) sw[s] *= inv;
  }
  __syncthreads();
  for (int d = tid; d < H_; d += 256) {
    float a = 0.f;
    for (int s = 0; s < S_; ++s) a += sw[s] * eb[(size_t)s * H_ + d];
    ctx_bf[(size_t)b * H_ + d] = f2bf(a);
  }
}

// dst[k*N + n] = bf16(src[n*lds + koff + k])  (transpose + convert weights)
__global__ void cd_transcvt(const float* __restrict__ src, int lds, int koff,
                            unsigned short* __restrict__ dst, int K, int N) {
  int n = blockIdx.x * blockDim.x + threadIdx.x;
  int k = blockIdx.y;
  if (n < N) dst[(size_t)k * N + n] = f2bf(src[(size_t)n * lds + koff + k]);
}

// X[r=t*B+b][:] = bf16(emb[tgt[b,t]][:])   (time-major packed decoder inputs)
__global__ void cd_gather_emb(const float* __restrict__ emb,
                              const int* __restrict__ tgt,
                              unsigned short* __restrict__ X) {
  int r = blockIdx.x, t = r >> 7, b = r & 127;
  int tok = tgt[b * T_ + t];
  const float* e = emb + (size_t)tok * WD_;
  unsigned short* xr = X + (size_t)r * WD_;
  for (int d = threadIdx.x; d < WD_; d += 256) xr[d] = f2bf(e[d]);
}

__global__ void cd_bsum(const float* a, const float* b, float* o, int n) {
  int i = blockIdx.x * blockDim.x + threadIdx.x;
  if (i < n) o[i] = a[i] + b[i];
}

__global__ void cd_init_hc(const float* __restrict__ h0,
                           unsigned short* __restrict__ hbf,
                           float* __restrict__ c) {
  int i = blockIdx.x * blockDim.x + threadIdx.x;  // B*H
  hbf[i] = f2bf(h0[i]);
  c[i] = 0.f;
}

// gates (B x 4H, torch order i,f,g,o) -> c, h(bf16), Hs[t] slice
__global__ void cd_lstm_cell(const float* __restrict__ gates,
                             float* __restrict__ c,
                             unsigned short* __restrict__ hbf,
                             unsigned short* __restrict__ HsBf, int t) {
  int i = blockIdx.x * blockDim.x + threadIdx.x;  // B*H
  int b = i >> 9, d = i & (H_ - 1);
  const float* g = gates + (size_t)b * G_;
  float ig = sigm(g[d]);
  float fg = sigm(g[H_ + d]);
  float gg = tanhf(g[2 * H_ + d]);
  float og = sigm(g[3 * H_ + d]);
  float cn = fg * c[i] + ig * gg;
  float hn = og * tanhf(cn);
  c[i] = cn;
  unsigned short hb = f2bf(hn);
  hbf[i] = hb;
  HsBf[(size_t)t * (B_ * H_) + i] = hb;
}

// argmax over V per (b,t); logits already in (B,T-1,V) order; first-max tiebreak
__global__ __launch_bounds__(256) void cd_argmax(const float* __restrict__ logits,
                                                 float* __restrict__ preds) {
  __shared__ float bv[256];
  __shared__ int   bi[256];
  int p = blockIdx.x, tid = threadIdx.x;
  const float* L = logits + (size_t)p * V_;
  float best = -3.4e38f; int bidx = V_;
  for (int v = tid; v < V_; v += 256) {
    float x = L[v];
    if (x > best) { best = x; bidx = v; }
  }
  bv[tid] = best; bi[tid] = bidx;
  __syncthreads();
  for (int s = 128; s > 0; s >>= 1) {
    if (tid < s) {
      if (bv[tid + s] > bv[tid] || (bv[tid + s] == bv[tid] && bi[tid + s] < bi[tid])) {
        bv[tid] = bv[tid + s]; bi[tid] = bi[tid + s];
      }
    }
    __syncthreads();
  }
  if (tid == 0) preds[p] = (float)bi[0];
}

extern "C" void kernel_launch(void* const* d_in, const int* in_sizes, int n_in,
                              void* d_out, int out_size, void* d_ws, size_t ws_size,
                              hipStream_t stream) {
  const float* h0     = (const float*)d_in[0];   // (1,B,H)
  const float* enc    = (const float*)d_in[1];   // (B,S,H)
  const int*   tgt    = (const int*)  d_in[2];   // (B,T)
  const float* emb    = (const float*)d_in[4];   // (V,WD)
  const float* fc1_w  = (const float*)d_in[5];
  const float* fc2_w  = (const float*)d_in[7];
  const float* fc3_w  = (const float*)d_in[9];
  const float* fc4_w  = (const float*)d_in[11];
  const float* attn_w = (const float*)d_in[13];
  const float* w_ih   = (const float*)d_in[14];  // (4H, WD+H)
  const float* w_hh   = (const float*)d_in[15];  // (4H, H)
  const float* b_ih   = (const float*)d_in[16];
  const float* b_hh   = (const float*)d_in[17];
  const float* out_w  = (const float*)d_in[18];  // (V, H)
  const float* out_b  = (const float*)d_in[19];
  float* out = (float*)d_out;                    // logits (B,T-1,V) then preds (B,T-1)

  const int ROWS = TM1 * B_;                     // 3456 time-major rows

  uintptr_t p = (uintptr_t)d_ws;
  auto take = [&](size_t bytes) -> uintptr_t {
    uintptr_t r = p; p += (bytes + 255) & ~(size_t)255; return r;
  };
  float* t3    = (float*)take(H_ * 4);
  float* t2    = (float*)take(H_ * 4);
  float* t1    = (float*)take(H_ * 4);
  float* t0e   = (float*)take(H_ * 4);
  float* bsum  = (float*)take(G_ * 4);
  float* gc    = (float*)take((size_t)B_ * G_ * 4);
  float* gates = (float*)take((size_t)B_ * G_ * 4);
  float* cbuf  = (float*)take((size_t)B_ * H_ * 4);
  float* Gx    = (float*)take((size_t)ROWS * G_ * 4);
  unsigned short* ctxbf = (unsigned short*)take((size_t)B_ * H_ * 2);
  unsigned short* hbf   = (unsigned short*)take((size_t)B_ * H_ * 2);
  unsigned short* Xemb  = (unsigned short*)take((size_t)ROWS * WD_ * 2);
  unsigned short* Wihx  = (unsigned short*)take((size_t)WD_ * G_ * 2);
  unsigned short* Wihc  = (unsigned short*)take((size_t)H_ * G_ * 2);
  unsigned short* Whh   = (unsigned short*)take((size_t)H_ * G_ * 2);
  unsigned short* Wout  = (unsigned short*)take((size_t)H_ * V_ * 2);
  unsigned short* HsBf  = (unsigned short*)take((size_t)ROWS * H_ * 2);

  // 1) collapse the linear attention stack into a single H-vector t0e
  cd_colvec<<<2, 256, 0, stream>>>(fc4_w, H_, attn_w, H_, t3, H_);
  cd_colvec<<<2, 256, 0, stream>>>(fc3_w, H_, t3, H_, t2, H_);
  cd_colvec<<<2, 256, 0, stream>>>(fc2_w, H_, t2, H_, t1, H_);
  cd_colvec<<<2, 256, 0, stream>>>(fc1_w, 2 * H_, t1, H_, t0e, H_);  // enc half only

  // 2) step-invariant attention context (softmax(h·...) term cancels)
  cd_attn_ctx<<<B_, 256, 0, stream>>>(enc, t0e, ctxbf);

  // 3) weight transpose+convert to bf16 (K x N, row-major for the GEMM B operand)
  cd_transcvt<<<dim3(G_ / 256, WD_), 256, 0, stream>>>(w_ih, LI_, 0,    Wihx, WD_, G_);
  cd_transcvt<<<dim3(G_ / 256, H_ ), 256, 0, stream>>>(w_ih, LI_, WD_,  Wihc, H_, G_);
  cd_transcvt<<<dim3(G_ / 256, H_ ), 256, 0, stream>>>(w_hh, H_,  0,    Whh,  H_, G_);
  cd_transcvt<<<dim3((V_ + 255) / 256, H_), 256, 0, stream>>>(out_w, H_, 0, Wout, H_, V_);

  // 4) pack all 27 steps of target embeddings (time-major)
  cd_gather_emb<<<ROWS, 256, 0, stream>>>(emb, tgt, Xemb);
  cd_bsum<<<G_ / 256, 256, 0, stream>>>(b_ih, b_hh, bsum, G_);

  // 5) big input-side gate GEMM: Gx = Xemb(3456x1024) @ Wihx(1024x2048)
  cd_gemm_bf16<<<dim3(G_ / TNT, ROWS / TMT), 256, 0, stream>>>(
      Xemb, WD_, Wihx, G_, Gx, nullptr, nullptr, nullptr, G_, WD_, 0, 0);

  // 6) context-side gates + biases (constant over steps): gc = ctx @ Wihc + bsum
  cd_gemm_bf16<<<dim3(G_ / TNT, 1), 256, 0, stream>>>(
      ctxbf, H_, Wihc, G_, gc, nullptr, nullptr, bsum, G_, H_, 0, 0);

  // 7) sequential LSTM recurrence (only h @ w_hh^T remains per step)
  cd_init_hc<<<(B_ * H_) / 256, 256, 0, stream>>>(h0, hbf, cbuf);
  for (int t = 0; t < TM1; ++t) {
    cd_gemm_bf16<<<dim3(G_ / TNT, 1), 256, 0, stream>>>(
        hbf, H_, Whh, G_, gates, Gx + (size_t)t * B_ * G_, gc, nullptr, G_, H_, 0, 0);
    cd_lstm_cell<<<(B_ * H_) / 256, 256, 0, stream>>>(gates, cbuf, hbf, HsBf, t);
  }

  // 8) batched logits GEMM, epilogue writes d_out directly in (B,T-1,V) order
  cd_gemm_bf16<<<dim3(V_ / TNT, ROWS / TMT), 256, 0, stream>>>(
      HsBf, H_, Wout, V_, out, nullptr, nullptr, out_b, V_, H_, B_, TM1);

  // 9) argmax -> preds appended after logits
  cd_argmax<<<ROWS, 256, 0, stream>>>(out, out + (size_t)ROWS * V_);
}
